// ConvAttention2d_49538152792395
// MI455X (gfx1250) — compile-verified, matrix operand order checked by reading, not run on hardware
//
#include <hip/hip_runtime.h>
#include <hip/hip_bf16.h>

// ---------------------------------------------------------------------------
// ConvAttention2d for MI455X (gfx1250, wave32, WMMA).
// Memory-bound (~17.5 GFLOP vs ~200MB @23.3TB/s). Pipeline:
//   0) one-shot bf16 pre-conversion: x -> xT[p][c] (LDS transpose), W -> bf16
//   1) QKV GEMM: all-bf16 operands, tiles filled with CDNA5 async DMA
//      (global_load_async_to_lds_b128, ASYNCcnt), math on v_wmma_16x16x32_bf16
//   2) 3x3 windowed softmax-attention: one wave32 per (pixel, head)
//   3) output GEMM + bias, async tile fills, f32 scatter to [b,c,H,W]
// f32->bf16 uses v_add + v_perm_b32 (3 VALU per packed pair).
// ---------------------------------------------------------------------------

typedef __attribute__((ext_vector_type(16))) __bf16 v16bf;
typedef __attribute__((ext_vector_type(8)))  float  v8f;

union FragAB { int4 q[2]; v16bf v; };          // 32B: one 16x32 bf16 operand fragment
union AccF   { v8f v; float f[8]; };           // 16x16 f32 accumulator fragment

#define BATCH   16
#define CIN     256
#define HW      1024            // 32*32
#define NPIX    (BATCH * HW)    // 16384
#define INNER   512             // heads*dh
#define QKV_N   1536            // q+k+v columns
#define SA      40              // LDS row stride in bf16 elems (pad, 16B-align ok)

// f32x2 -> packed bf16x2: round-half-up + byte-perm of the two high halves.
__device__ __forceinline__ unsigned pack_bf2(float a, float b) {
    unsigned ua = __float_as_uint(a) + 0x8000u;
    unsigned ub = __float_as_uint(b) + 0x8000u;
    return __builtin_amdgcn_perm(ua, ub, 0x03020706u);  // {ub.h, ua.h}
}
__device__ __forceinline__ float bf2f(unsigned short h) {
    return __uint_as_float(((unsigned)h) << 16);
}
// per-lane async global->LDS 16B copy (CDNA5; generic LDS ptr low 32b = LDS addr)
__device__ __forceinline__ void async_b128(const void* lds, const void* gptr) {
    unsigned           ldsa = (unsigned)(unsigned long long)lds;
    unsigned long long ga   = (unsigned long long)gptr;
    asm volatile("global_load_async_to_lds_b128 %0, %1, off"
                 :: "v"(ldsa), "v"(ga) : "memory");
}

// ===========================================================================
// Pre-pass A: x[b][c][n] f32 -> xT[b*HW+n][c] bf16 via 64x64 LDS transpose.
// grid (HW/64, CIN/64, BATCH), 256 thr.
// ===========================================================================
__global__ __launch_bounds__(256)
void cvt_x(const float* __restrict__ x, unsigned short* __restrict__ xT) {
    __shared__ unsigned short sT[64 * 72];
    const int tid = threadIdx.x;
    const int nt = blockIdx.x, ct = blockIdx.y, bb = blockIdx.z;
    const float* src = x + (size_t)bb * CIN * HW + (size_t)ct * 64 * HW + nt * 64;
#pragma unroll
    for (int i = 0; i < 4; ++i) {
        int idx = tid + 256 * i;                // 0..1023
        int r   = idx >> 4;                     // channel row 0..63
        int c4  = idx & 15;                     // float4 within 64 pixels
        float4 f = *(const float4*)(src + (size_t)r * HW + c4 * 4);
        unsigned u01 = pack_bf2(f.x, f.y);
        unsigned u23 = pack_bf2(f.z, f.w);
        int nl = c4 * 4;
        sT[(nl + 0) * 72 + r] = (unsigned short)u01;
        sT[(nl + 1) * 72 + r] = (unsigned short)(u01 >> 16);
        sT[(nl + 2) * 72 + r] = (unsigned short)u23;
        sT[(nl + 3) * 72 + r] = (unsigned short)(u23 >> 16);
    }
    __syncthreads();
    unsigned short* dst = xT + ((size_t)bb * HW + nt * 64) * CIN + ct * 64;
#pragma unroll
    for (int i = 0; i < 2; ++i) {
        int idx = tid + 256 * i;                // 0..511
        int nl  = idx >> 3;
        int c8  = idx & 7;
        *(int4*)(dst + (size_t)nl * CIN + c8 * 8) = *(const int4*)&sT[nl * 72 + c8 * 8];
    }
}

// Pre-pass B: flat f32 -> bf16 (weights). n4 = elements/4.
__global__ __launch_bounds__(256)
void cvt_bf16(const float* __restrict__ src, unsigned short* __restrict__ dst, int n4) {
    int i = blockIdx.x * 256 + threadIdx.x;
    if (i < n4) {
        float4 f = ((const float4*)src)[i];
        uint2 u;
        u.x = pack_bf2(f.x, f.y);
        u.y = pack_bf2(f.z, f.w);
        ((uint2*)dst)[i] = u;
    }
}

// ===========================================================================
// Kernel 1: QKV projection GEMM, all-bf16 operands.
//   A = xT [16384][256], B = Wc [1536][256], D = qkv bf16 [16384][1536]
// Block = 256 thr (8 waves). Tile 128(M) x 128(N), K-step 32. Async LDS fills.
// ===========================================================================
__global__ __launch_bounds__(256)
void qkv_gemm(const unsigned short* __restrict__ xT,
              const unsigned short* __restrict__ Wc,
              unsigned short* __restrict__ qkv) {
    __shared__ unsigned short sA[128 * SA];     // [pixel][k]
    __shared__ unsigned short sB[128 * SA];     // [outcol][k]

    const int tid  = threadIdx.x;
    const int lane = tid & 31;
    const int wid  = tid >> 5;
    const int wm   = wid >> 1;                  // 0..3 -> M offset wm*32
    const int wn   = wid & 1;                   // 0..1 -> N offset wn*64
    const int lmod = lane & 15;
    const int kh   = (lane >> 4) * 8;           // bf16-K half select per ISA layout

    const int p0 = blockIdx.x * 128;
    const int n0 = blockIdx.y * 128;

    const int r  = tid >> 2;                    // loader row 0..63 (+64 on i=1)
    const int c4 = tid & 3;                     // 16B chunk within 64B k-slice

    AccF acc[2][4];
#pragma unroll
    for (int mi = 0; mi < 2; ++mi)
#pragma unroll
        for (int ni = 0; ni < 4; ++ni) acc[mi][ni].v = (v8f){};

    for (int kt = 0; kt < CIN / 32; ++kt) {
        const int c0 = kt * 32;
#pragma unroll
        for (int i = 0; i < 2; ++i) {
            int rr = r + 64 * i;
            async_b128(&sA[rr * SA + c4 * 8],
                       xT + (size_t)(p0 + rr) * CIN + c0 + c4 * 8);
            async_b128(&sB[rr * SA + c4 * 8],
                       Wc + (size_t)(n0 + rr) * CIN + c0 + c4 * 8);
        }
        if (kt + 1 < CIN / 32) {                 // gfx1250 global_prefetch_b8
            __builtin_prefetch(xT + (size_t)(p0 + r) * CIN + c0 + 32, 0, 1);
            __builtin_prefetch(Wc + (size_t)(n0 + r) * CIN + c0 + 32, 0, 1);
        }
        asm volatile("s_wait_asynccnt 0x0" ::: "memory");
        __syncthreads();

        FragAB fa[2], fb[4];
#pragma unroll
        for (int mi = 0; mi < 2; ++mi) {
            const unsigned short* pa = &sA[(wm * 32 + mi * 16 + lmod) * SA];
            fa[mi].q[0] = *(const int4*)(pa + kh);
            fa[mi].q[1] = *(const int4*)(pa + kh + 16);
        }
#pragma unroll
        for (int ni = 0; ni < 4; ++ni) {
            const unsigned short* pb = &sB[(wn * 64 + ni * 16 + lmod) * SA];
            fb[ni].q[0] = *(const int4*)(pb + kh);
            fb[ni].q[1] = *(const int4*)(pb + kh + 16);
        }
#pragma unroll
        for (int mi = 0; mi < 2; ++mi)
#pragma unroll
            for (int ni = 0; ni < 4; ++ni)
                acc[mi][ni].v = __builtin_amdgcn_wmma_f32_16x16x32_bf16(
                    false, fa[mi].v, false, fb[ni].v,
                    (short)0, acc[mi][ni].v, false, false);
        __syncthreads();
    }

    // epilogue: D layout (ISA): lanes 0-15 M=0..7, lanes 16-31 M=8..15
    const int prow0 = p0 + wm * 32 + (lane >> 4) * 8;
#pragma unroll
    for (int mi = 0; mi < 2; ++mi)
#pragma unroll
        for (int ni = 0; ni < 4; ++ni) {
            int ncol = n0 + wn * 64 + ni * 16 + lmod;
#pragma unroll
            for (int rr = 0; rr < 8; rr += 2) {
                unsigned u = pack_bf2(acc[mi][ni].f[rr], acc[mi][ni].f[rr + 1]);
                qkv[(size_t)(prow0 + mi * 16 + rr)     * QKV_N + ncol] = (unsigned short)u;
                qkv[(size_t)(prow0 + mi * 16 + rr + 1) * QKV_N + ncol] = (unsigned short)(u >> 16);
            }
        }
}

// ===========================================================================
// Kernel 2: 3x3 windowed attention. One wave32 per (pixel, head).
// Lane holds d-elements {2L, 2L+1}; dots via butterfly shuffle reduction.
// Zero-padded borders => dot = 0 there (matches reference unfold semantics).
// ===========================================================================
__global__ __launch_bounds__(256)
void win_attn(const unsigned short* __restrict__ qkv,
              unsigned short* __restrict__ Ob,
              float* __restrict__ attn_out) {
    const int tid  = threadIdx.x;
    const int lane = tid & 31;
    const int wid  = blockIdx.x * 8 + (tid >> 5);   // 0 .. 131071
    const int head = wid & 7;
    const int p    = wid >> 3;
    const int bb   = p >> 10;
    const int n    = p & 1023;
    const int yy0  = n >> 5;
    const int xx0  = n & 31;
    const int d0   = 2 * lane;
    const float scale = 0.125f;                     // 64^-0.5

    unsigned qu = *(const unsigned*)(qkv + (size_t)p * QKV_N + head * 64 + d0);
    float q0 = bf2f((unsigned short)(qu & 0xFFFF));
    float q1 = bf2f((unsigned short)(qu >> 16));

    float dot[9];
#pragma unroll
    for (int w = 0; w < 9; ++w) {
        int dy = w / 3 - 1, dx = w % 3 - 1;
        int yy = yy0 + dy, xx = xx0 + dx;
        bool valid = ((unsigned)yy < 32u) & ((unsigned)xx < 32u);
        float part = 0.f;
        if (valid) {
            size_t p2 = (size_t)bb * HW + yy * 32 + xx;
            unsigned ku = *(const unsigned*)(qkv + p2 * QKV_N + 512 + head * 64 + d0);
            part = q0 * bf2f((unsigned short)(ku & 0xFFFF)) +
                   q1 * bf2f((unsigned short)(ku >> 16));
        }
#pragma unroll
        for (int m = 16; m > 0; m >>= 1) part += __shfl_xor(part, m, 32);
        dot[w] = part * scale;
    }

    float mx = dot[0];
#pragma unroll
    for (int w = 1; w < 9; ++w) mx = fmaxf(mx, dot[w]);
    float e[9], s = 0.f;
#pragma unroll
    for (int w = 0; w < 9; ++w) { e[w] = __expf(dot[w] - mx); s += e[w]; }
    float inv = 1.f / s;

    float o0 = 0.f, o1 = 0.f;
#pragma unroll
    for (int w = 0; w < 9; ++w) {
        int dy = w / 3 - 1, dx = w % 3 - 1;
        int yy = yy0 + dy, xx = xx0 + dx;
        if (((unsigned)yy < 32u) & ((unsigned)xx < 32u)) {
            size_t p2 = (size_t)bb * HW + yy * 32 + xx;
            unsigned vu = *(const unsigned*)(qkv + p2 * QKV_N + 1024 + head * 64 + d0);
            float a = e[w] * inv;
            o0 += a * bf2f((unsigned short)(vu & 0xFFFF));
            o1 += a * bf2f((unsigned short)(vu >> 16));
        }
    }

    ((unsigned*)Ob)[(size_t)p * (INNER / 2) + head * 32 + lane] = pack_bf2(o0, o1);

    if (lane == 0) {
        size_t base = (((size_t)bb * 8 + head) * HW + n) * 9;
#pragma unroll
        for (int w = 0; w < 9; ++w) attn_out[base + w] = e[w] * inv;
    }
}

// ===========================================================================
// Kernel 3: output projection GEMM + bias, scatter into [b,c,H,W].
//   A = Ob bf16 [16384][512], B = Wob bf16 [256][512]. Async LDS fills.
// ===========================================================================
__global__ __launch_bounds__(256)
void out_gemm(const unsigned short* __restrict__ Ob,
              const unsigned short* __restrict__ Wob,
              const float* __restrict__ bo, float* __restrict__ out) {
    __shared__ unsigned short sA[128 * SA];
    __shared__ unsigned short sB[128 * SA];

    const int tid  = threadIdx.x;
    const int lane = tid & 31;
    const int wid  = tid >> 5;
    const int wm   = wid >> 1;
    const int wn   = wid & 1;
    const int lmod = lane & 15;
    const int kh   = (lane >> 4) * 8;

    const int p0 = blockIdx.x * 128;
    const int n0 = blockIdx.y * 128;
    const int r  = tid >> 2;
    const int c4 = tid & 3;

    AccF acc[2][4];
#pragma unroll
    for (int mi = 0; mi < 2; ++mi)
#pragma unroll
        for (int ni = 0; ni < 4; ++ni) acc[mi][ni].v = (v8f){};

    for (int kt = 0; kt < INNER / 32; ++kt) {
        const int c0 = kt * 32;
#pragma unroll
        for (int i = 0; i < 2; ++i) {
            int rr = r + 64 * i;
            async_b128(&sA[rr * SA + c4 * 8],
                       Ob + (size_t)(p0 + rr) * INNER + c0 + c4 * 8);
            async_b128(&sB[rr * SA + c4 * 8],
                       Wob + (size_t)((n0 + rr) & 255) * INNER + c0 + c4 * 8);
        }
        if (kt + 1 < INNER / 32) {
            __builtin_prefetch(Ob + (size_t)(p0 + r) * INNER + c0 + 32, 0, 1);
            __builtin_prefetch(Wob + (size_t)((n0 + r) & 255) * INNER + c0 + 32, 0, 1);
        }
        asm volatile("s_wait_asynccnt 0x0" ::: "memory");
        __syncthreads();

        FragAB fa[2], fb[4];
#pragma unroll
        for (int mi = 0; mi < 2; ++mi) {
            const unsigned short* pa = &sA[(wm * 32 + mi * 16 + lmod) * SA];
            fa[mi].q[0] = *(const int4*)(pa + kh);
            fa[mi].q[1] = *(const int4*)(pa + kh + 16);
        }
#pragma unroll
        for (int ni = 0; ni < 4; ++ni) {
            const unsigned short* pb = &sB[(wn * 64 + ni * 16 + lmod) * SA];
            fb[ni].q[0] = *(const int4*)(pb + kh);
            fb[ni].q[1] = *(const int4*)(pb + kh + 16);
        }
#pragma unroll
        for (int mi = 0; mi < 2; ++mi)
#pragma unroll
            for (int ni = 0; ni < 4; ++ni)
                acc[mi][ni].v = __builtin_amdgcn_wmma_f32_16x16x32_bf16(
                    false, fa[mi].v, false, fb[ni].v,
                    (short)0, acc[mi][ni].v, false, false);
        __syncthreads();
    }

    const int prow0 = p0 + wm * 32 + (lane >> 4) * 8;
#pragma unroll
    for (int mi = 0; mi < 2; ++mi)
#pragma unroll
        for (int ni = 0; ni < 4; ++ni) {
            int ocol = n0 + wn * 64 + ni * 16 + lmod;
            float bias = bo[ocol];
#pragma unroll
            for (int rr = 0; rr < 8; ++rr) {
                int p  = prow0 + mi * 16 + rr;
                int bb = p >> 10;
                int nn = p & 1023;
                out[(size_t)bb * CIN * HW + (size_t)ocol * HW + nn] =
                    acc[mi][ni].f[rr] + bias;
            }
        }
}

// ===========================================================================
extern "C" void kernel_launch(void* const* d_in, const int* in_sizes, int n_in,
                              void* d_out, int out_size, void* d_ws, size_t ws_size,
                              hipStream_t stream) {
    const float* x   = (const float*)d_in[0];
    const float* Wq  = (const float*)d_in[1];
    const float* Wkv = (const float*)d_in[2];
    const float* Wo  = (const float*)d_in[3];
    const float* bo  = (const float*)d_in[4];

    float* outp = (float*)d_out;                         // [16,256,32,32]
    float* attn = outp + (size_t)BATCH * CIN * HW;       // [16,8,1024,9]

    char* ws = (char*)d_ws;
    unsigned short* qkv = (unsigned short*)ws;                               // 48 MB
    unsigned short* Ob  = (unsigned short*)(ws + (size_t)NPIX * QKV_N * 2);  // 16 MB
    unsigned short* xT  = (unsigned short*)(ws + (size_t)NPIX * (QKV_N + INNER) * 2); // 8 MB
    unsigned short* Wc  = xT + (size_t)NPIX * CIN;                           // 768 KB
    unsigned short* Wob = Wc + (size_t)QKV_N * CIN;                          // 256 KB

    // pre-convert to bf16 (x also transposed to [p][c])
    cvt_x<<<dim3(HW / 64, CIN / 64, BATCH), 256, 0, stream>>>(x, xT);
    cvt_bf16<<<dim3((INNER * CIN / 4) / 256), 256, 0, stream>>>(Wq, Wc, INNER * CIN / 4);
    cvt_bf16<<<dim3((2 * INNER * CIN / 4) / 256), 256, 0, stream>>>(Wkv, Wc + (size_t)INNER * CIN,
                                                                    2 * INNER * CIN / 4);
    cvt_bf16<<<dim3((CIN * INNER / 4) / 256), 256, 0, stream>>>(Wo, Wob, CIN * INNER / 4);

    qkv_gemm<<<dim3(NPIX / 128, QKV_N / 128), 256, 0, stream>>>(xT, Wc, qkv);
    win_attn<<<dim3(NPIX), 256, 0, stream>>>(qkv, Ob, attn);
    out_gemm<<<dim3(NPIX / 128, CIN / 128), 256, 0, stream>>>(Ob, Wob, bo, outp);
}
